// TransBlock_6871947674090
// MI455X (gfx1250) — matrix operand order repaired and checked by reading, not compiled, not run on hardware
//
#include <hip/hip_runtime.h>

// ---------------- problem constants ----------------
#define BATCH 4
#define SEQ   2048
#define T_TOK (BATCH * SEQ)      // 8192 tokens
#define EMB   1024
#define NHEAD 16
#define DHID  64
#define DFF   4096

// ---------------- WMMA fragment types ----------------
typedef __attribute__((ext_vector_type(16))) __bf16        v16bf;
typedef __attribute__((ext_vector_type(8)))  float         v8f;
typedef __attribute__((ext_vector_type(8)))  unsigned short u16x8;

union Frag {
    u16x8          h[2];
    unsigned short s[16];
    v16bf          v;
};

__device__ __forceinline__ unsigned short f2bf(float f) {
    union { float f; unsigned int u; } c; c.f = f;
    unsigned int r = c.u + 0x7FFFu + ((c.u >> 16) & 1u);   // round-to-nearest-even
    return (unsigned short)(r >> 16);
}

__device__ __forceinline__ v8f wmma_bf16(v16bf a, v16bf b, v8f c) {
    return __builtin_amdgcn_wmma_f32_16x16x32_bf16(false, a, false, b, (short)0, c,
                                                   false, false);
}

// A-matrix 16x32 bf16 fragment in this file's contiguous k order (B uses the same; not the ISA table order):
// lane l (m = l&15, hi = l>>4) holds K = {kk+hi*16 .. +15}
__device__ __forceinline__ v16bf load_a_frag(const unsigned short* row, int kk, int hi) {
    Frag f;
    f.h[0] = *(const u16x8*)(row + kk + hi * 16);
    f.h[1] = *(const u16x8*)(row + kk + hi * 16 + 8);
    return f.v;
}

// ---------------- fp32 -> bf16 cast ----------------
__global__ __launch_bounds__(256) void cast_bf16_kernel(const float* __restrict__ in,
                                                        unsigned short* __restrict__ out,
                                                        int n) {
    int i = blockIdx.x * 256 + threadIdx.x;
    if (i < n) out[i] = f2bf(in[i]);
}

// ---------------- LayerNorm (block per row) -> bf16 ----------------
__global__ __launch_bounds__(256) void ln_bf16_kernel(const float* __restrict__ x,
                                                      const float* __restrict__ g,
                                                      const float* __restrict__ b,
                                                      unsigned short* __restrict__ out,
                                                      int E) {
    __shared__ float red[2][8];
    int row = blockIdx.x;
    const float* xr = x + (size_t)row * E;
    float s = 0.f, s2 = 0.f;
    for (int i = threadIdx.x; i < E; i += 256) {
        float v = xr[i];
        s += v; s2 += v * v;
    }
    for (int m = 16; m >= 1; m >>= 1) {
        s  += __shfl_xor(s,  m, 32);
        s2 += __shfl_xor(s2, m, 32);
    }
    int wave = threadIdx.x >> 5, lane = threadIdx.x & 31;
    if (lane == 0) { red[0][wave] = s; red[1][wave] = s2; }
    __syncthreads();
    if (wave == 0) {
        s = red[0][lane & 7]; s2 = red[1][lane & 7];
        for (int m = 4; m >= 1; m >>= 1) {
            s  += __shfl_xor(s,  m, 8);
            s2 += __shfl_xor(s2, m, 8);
        }
        if (lane == 0) { red[0][0] = s; red[1][0] = s2; }
    }
    __syncthreads();
    float mu  = red[0][0] / (float)E;
    float var = red[1][0] / (float)E - mu * mu;
    float inv = rsqrtf(var + 1e-5f);
    for (int i = threadIdx.x; i < E; i += 256) {
        float v = (xr[i] - mu) * inv * g[i] + b[i];
        out[(size_t)row * E + i] = f2bf(v);
    }
}

// ---------------- generic WMMA GEMM: Y = X[M,K] * W[N,K]^T ----------------
// wave = 16 rows x 128 cols, block = 8 waves = 128x128 tile.
// Register double-buffered; sched_barrier(0) pins each LD phase ahead of the
// MM phase it must overlap (prevents the scheduler from sinking the prefetch).
__global__ __launch_bounds__(256) void gemm_bf16_kernel(
    const unsigned short* __restrict__ X,   // [M,K] bf16
    const unsigned short* __restrict__ W,   // [N,K] bf16
    const float* __restrict__ bias,         // [N] or null
    const float* __restrict__ res,          // [M,N] fp32 or null
    float alpha, int M, int N, int K,       // K is a power of two (1024/4096)
    int relu, int out_is_bf16, int vt_mode,
    unsigned short* __restrict__ out_bf,
    float* __restrict__ out_f) {
    int lane = threadIdx.x & 31;
    int wave = threadIdx.x >> 5;
    int hi = lane >> 4, lo = lane & 15;
    int r0 = blockIdx.y * 128 + wave * 16;
    int n0 = blockIdx.x * 128;

    const unsigned short* arow = X + (size_t)(r0 + lo) * K;
    const unsigned short* brow[8];
#pragma unroll
    for (int nt = 0; nt < 8; nt++) brow[nt] = W + (size_t)(n0 + nt * 16 + lo) * K;

    v8f acc[8];
#pragma unroll
    for (int t = 0; t < 8; t++)
#pragma unroll
        for (int v = 0; v < 8; v++) acc[t][v] = 0.f;

    Frag a0, a1;
    Frag b0[8], b1[8];

    auto LD = [&](Frag& a, Frag(&bb)[8], int kk) {
        a.h[0] = *(const u16x8*)(arow + kk + hi * 16);
        a.h[1] = *(const u16x8*)(arow + kk + hi * 16 + 8);
#pragma unroll
        for (int nt = 0; nt < 8; nt++) {
            bb[nt].h[0] = *(const u16x8*)(brow[nt] + kk + hi * 16);
            bb[nt].h[1] = *(const u16x8*)(brow[nt] + kk + hi * 16 + 8);
        }
    };
    auto MM = [&](Frag& a, Frag(&bb)[8]) {
#pragma unroll
        for (int nt = 0; nt < 8; nt++) acc[nt] = wmma_bf16(a.v, bb[nt].v, acc[nt]);
    };

    const int Kmask = K - 1;
    LD(a0, b0, 0);
    for (int kk = 0; kk < K; kk += 64) {
        LD(a1, b1, kk + 32);                       // prefetch step k+1
        __builtin_amdgcn_sched_barrier(0);
        MM(a0, b0);                                // compute step k
        __builtin_amdgcn_sched_barrier(0);
        LD(a0, b0, (kk + 64) & Kmask);             // prefetch step k+2 (wraps on tail; unused)
        __builtin_amdgcn_sched_barrier(0);
        MM(a1, b1);                                // compute step k+1
        __builtin_amdgcn_sched_barrier(0);
    }

    // C layout: lane l holds col = l&15, rows v + 8*(l>>4)
#pragma unroll
    for (int nt = 0; nt < 8; nt++) {
        int col = n0 + nt * 16 + lo;
        float bc = bias ? bias[col] : 0.f;
#pragma unroll
        for (int v = 0; v < 8; v++) {
            int row = r0 + v + 8 * hi;
            float val = acc[nt][v] * alpha + bc;
            if (relu) val = fmaxf(val, 0.f);
            if (res) val += res[(size_t)row * N + col];
            if (vt_mode) {
                // store V transposed: vt[((b*16+h)*64 + d) * SEQ + s]
                int bb = row >> 11, ss = row & (SEQ - 1);
                size_t idx = ((size_t)(bb * NHEAD + (col >> 6)) * DHID + (col & 63)) * SEQ + ss;
                out_bf[idx] = f2bf(val);
            } else if (out_is_bf16) {
                out_bf[(size_t)row * N + col] = f2bf(val);
            } else {
                out_f[(size_t)row * N + col] = val;
            }
        }
    }
}

// ---------------- flash attention: one wave per (b, h, 16-query tile) ----------------
// Pipelined: V fragments + next key tile's K fragments issued right after the
// QK^T WMMAs and pinned there (latency hidden under the softmax VALU work).
__global__ __launch_bounds__(256) void flash_attn_kernel(
    const unsigned short* __restrict__ Q,   // [T,E] bf16, pre-scaled by 1/sqrt(d)
    const unsigned short* __restrict__ Kb,  // [T,E] bf16
    const unsigned short* __restrict__ VT,  // [B*H*DHID, SEQ] bf16 (transposed V)
    unsigned short* __restrict__ O) {       // [T,E] bf16
    __shared__ __align__(16) unsigned short pbuf[8][16 * 32];

    int lane = threadIdx.x & 31;
    int wave = threadIdx.x >> 5;
    int hi = lane >> 4, lo = lane & 15;

    const int qtiles = SEQ / 16;                       // 128
    int wid = blockIdx.x * 8 + wave;
    int b   = wid / (NHEAD * qtiles);
    int rem = wid % (NHEAD * qtiles);
    int h   = rem / qtiles;
    int q0  = (rem % qtiles) * 16;

    const unsigned short* qrow = Q + ((size_t)(b * SEQ + q0 + lo)) * EMB + h * DHID;
    v16bf qa0 = load_a_frag(qrow, 0, hi);    // d = 0..31
    v16bf qa1 = load_a_frag(qrow, 32, hi);   // d = 32..63

    const unsigned short* kbase = Kb + (size_t)(b * SEQ) * EMB + h * DHID;      // + key*EMB
    const unsigned short* vbase = VT + ((size_t)(b * NHEAD + h) * DHID) * SEQ;  // + d*SEQ + s

    float m[8], ssum[8];
    v8f acc[4];
#pragma unroll
    for (int v = 0; v < 8; v++) { m[v] = -1e30f; ssum[v] = 0.f; }
#pragma unroll
    for (int t = 0; t < 4; t++)
#pragma unroll
        for (int v = 0; v < 8; v++) acc[t][v] = 0.f;

    unsigned short* pl = &pbuf[wave][0];

    // kf[ct*2 + dstep]: B fragments of K^T for one 32-key tile
    auto loadK4 = [&](Frag(&kf)[4], int k0) {
#pragma unroll
        for (int ct = 0; ct < 2; ct++) {
            const unsigned short* krow = kbase + (size_t)(k0 + ct * 16 + lo) * EMB;
            kf[ct * 2 + 0].h[0] = *(const u16x8*)(krow + hi * 16);
            kf[ct * 2 + 0].h[1] = *(const u16x8*)(krow + hi * 16 + 8);
            kf[ct * 2 + 1].h[0] = *(const u16x8*)(krow + 32 + hi * 16);
            kf[ct * 2 + 1].h[1] = *(const u16x8*)(krow + 32 + hi * 16 + 8);
        }
    };

    auto step = [&](Frag(&kf)[4], int k0, bool pre, Frag(&kn)[4], int knext) {
        // ---- S = Q K^T : two 16-key column tiles, K-dim = 64 (2 WMMA steps) ----
        v8f sc[2];
#pragma unroll
        for (int ct = 0; ct < 2; ct++) {
            v8f c;
#pragma unroll
            for (int v = 0; v < 8; v++) c[v] = 0.f;
            c = wmma_bf16(qa0, kf[ct * 2 + 0].v, c);
            c = wmma_bf16(qa1, kf[ct * 2 + 1].v, c);
            sc[ct] = c;
        }

        // ---- issue V-fragment loads NOW (latency hidden under softmax VALU) ----
        Frag vf[4];
#pragma unroll
        for (int t = 0; t < 4; t++) {
            const unsigned short* vrow = vbase + (size_t)(t * 16 + lo) * SEQ + k0;
            vf[t].h[0] = *(const u16x8*)(vrow + hi * 16);
            vf[t].h[1] = *(const u16x8*)(vrow + hi * 16 + 8);
        }
        // ---- issue next key tile's K fragments ----
        if (pre) loadK4(kn, knext);
        __builtin_amdgcn_sched_barrier(0);   // keep the loads above the softmax

        // ---- online softmax; row r = v + 8*hi lives across the 16 lanes of a half ----
#pragma unroll
        for (int v = 0; v < 8; v++) {
            float v0 = sc[0][v], v1 = sc[1][v];
            float rmax = fmaxf(v0, v1);
            for (int msk = 8; msk >= 1; msk >>= 1)
                rmax = fmaxf(rmax, __shfl_xor(rmax, msk, 16));
            float mn = fmaxf(m[v], rmax);
            float al = __expf(m[v] - mn);
            float p0 = __expf(v0 - mn);
            float p1 = __expf(v1 - mn);
            float rs = p0 + p1;
            for (int msk = 8; msk >= 1; msk >>= 1)
                rs += __shfl_xor(rs, msk, 16);
            ssum[v] = ssum[v] * al + rs;
            m[v] = mn;
#pragma unroll
            for (int t = 0; t < 4; t++) acc[t][v] *= al;
            // stage P in LDS (C layout -> memory row-major 16x32)
            int row = v + 8 * hi;
            pl[row * 32 + lo]      = f2bf(p0);
            pl[row * 32 + 16 + lo] = f2bf(p1);
        }
        asm volatile("s_wait_dscnt 0x0" ::: "memory");

        // re-read P as an A-matrix 16x32 fragment
        Frag pf;
        {
            const unsigned short* r = pl + lo * 32;
            pf.h[0] = *(const u16x8*)(r + hi * 16);
            pf.h[1] = *(const u16x8*)(r + hi * 16 + 8);
        }

        // ---- O += P * V : 4 output d-tiles (V fragments already arrived) ----
#pragma unroll
        for (int t = 0; t < 4; t++) acc[t] = wmma_bf16(pf.v, vf[t].v, acc[t]);
    };

    Frag kA[4], kB[4];
    loadK4(kA, 0);
    for (int k0 = 0; k0 < SEQ; k0 += 64) {        // 32 double-iterations
        step(kA, k0, true, kB, k0 + 32);
        step(kB, k0 + 32, k0 + 64 < SEQ, kA, k0 + 64);
    }

    // ---- normalize + store ----
#pragma unroll
    for (int t = 0; t < 4; t++)
#pragma unroll
        for (int v = 0; v < 8; v++) {
            int row = v + 8 * hi;
            float val = acc[t][v] / ssum[v];
            O[((size_t)(b * SEQ + q0 + row)) * EMB + h * DHID + t * 16 + lo] = f2bf(val);
        }
}

// ---------------- host orchestration ----------------
extern "C" void kernel_launch(void* const* d_in, const int* in_sizes, int n_in,
                              void* d_out, int out_size, void* d_ws, size_t ws_size,
                              hipStream_t stream) {
    const float* x    = (const float*)d_in[0];
    const float* Wq   = (const float*)d_in[1];
    const float* Wk   = (const float*)d_in[2];
    const float* Wv   = (const float*)d_in[3];
    const float* Wo   = (const float*)d_in[4];
    const float* bo   = (const float*)d_in[5];
    const float* ln1g = (const float*)d_in[6];
    const float* ln1b = (const float*)d_in[7];
    const float* ln2g = (const float*)d_in[8];
    const float* ln2b = (const float*)d_in[9];
    const float* W1   = (const float*)d_in[10];
    const float* b1   = (const float*)d_in[11];
    const float* W2   = (const float*)d_in[12];
    const float* b2   = (const float*)d_in[13];
    float* out = (float*)d_out;

    char* ws = (char*)d_ws;
    size_t off = 0;
    auto take = [&](size_t bytes) -> char* {
        char* p = ws + off;
        off = (off + bytes + 255) & ~(size_t)255;
        return p;
    };
    unsigned short* wqb  = (unsigned short*)take((size_t)EMB * EMB * 2);
    unsigned short* wkb  = (unsigned short*)take((size_t)EMB * EMB * 2);
    unsigned short* wvb  = (unsigned short*)take((size_t)EMB * EMB * 2);
    unsigned short* wob  = (unsigned short*)take((size_t)EMB * EMB * 2);
    unsigned short* w1b  = (unsigned short*)take((size_t)DFF * EMB * 2);
    unsigned short* w2b  = (unsigned short*)take((size_t)EMB * DFF * 2);
    unsigned short* xn   = (unsigned short*)take((size_t)T_TOK * EMB * 2);  // LN1 out, reused for LN2
    unsigned short* qb   = (unsigned short*)take((size_t)T_TOK * EMB * 2);
    unsigned short* kb   = (unsigned short*)take((size_t)T_TOK * EMB * 2);
    unsigned short* vt   = (unsigned short*)take((size_t)T_TOK * EMB * 2);
    unsigned short* attn = (unsigned short*)take((size_t)T_TOK * EMB * 2);
    float*          x1   = (float*)take((size_t)T_TOK * EMB * 4);
    unsigned short* hb   = (unsigned short*)take((size_t)T_TOK * DFF * 2);
    (void)ws_size; (void)in_sizes; (void)n_in; (void)out_size;

    // 1) weights -> bf16
    cast_bf16_kernel<<<(EMB * EMB + 255) / 256, 256, 0, stream>>>(Wq, wqb, EMB * EMB);
    cast_bf16_kernel<<<(EMB * EMB + 255) / 256, 256, 0, stream>>>(Wk, wkb, EMB * EMB);
    cast_bf16_kernel<<<(EMB * EMB + 255) / 256, 256, 0, stream>>>(Wv, wvb, EMB * EMB);
    cast_bf16_kernel<<<(EMB * EMB + 255) / 256, 256, 0, stream>>>(Wo, wob, EMB * EMB);
    cast_bf16_kernel<<<(DFF * EMB + 255) / 256, 256, 0, stream>>>(W1, w1b, DFF * EMB);
    cast_bf16_kernel<<<(EMB * DFF + 255) / 256, 256, 0, stream>>>(W2, w2b, EMB * DFF);

    // 2) LN1(x) -> xn (bf16)
    ln_bf16_kernel<<<T_TOK, 256, 0, stream>>>(x, ln1g, ln1b, xn, EMB);

    const float scaling = 0.125f;  // 1/sqrt(64)
    dim3 gEE(EMB / 128, T_TOK / 128);

    // 3) q = LN1(x) @ Wq^T * scaling   (bf16 out)
    gemm_bf16_kernel<<<gEE, 256, 0, stream>>>(xn, wqb, nullptr, nullptr, scaling,
                                              T_TOK, EMB, EMB, 0, 1, 0, qb, nullptr);
    // 4) k = LN1(x) @ Wk^T             (bf16 out)
    gemm_bf16_kernel<<<gEE, 256, 0, stream>>>(xn, wkb, nullptr, nullptr, 1.f,
                                              T_TOK, EMB, EMB, 0, 1, 0, kb, nullptr);
    // 5) v = LN1(x) @ Wv^T             (bf16 out, stored transposed per head)
    gemm_bf16_kernel<<<gEE, 256, 0, stream>>>(xn, wvb, nullptr, nullptr, 1.f,
                                              T_TOK, EMB, EMB, 0, 1, 1, vt, nullptr);

    // 6) flash attention -> attn (bf16)
    flash_attn_kernel<<<(BATCH * NHEAD * (SEQ / 16)) / 8, 256, 0, stream>>>(qb, kb, vt, attn);

    // 7) x1 = attn @ Wo^T + bo + x     (fp32 out)
    gemm_bf16_kernel<<<gEE, 256, 0, stream>>>(attn, wob, bo, x, 1.f,
                                              T_TOK, EMB, EMB, 0, 0, 0, nullptr, x1);

    // 8) LN2(x1) -> xn (bf16, buffer reuse)
    ln_bf16_kernel<<<T_TOK, 256, 0, stream>>>(x1, ln2g, ln2b, xn, EMB);

    // 9) h = relu(LN2 @ W1^T + b1)     (bf16 out)
    dim3 gFF1(DFF / 128, T_TOK / 128);
    gemm_bf16_kernel<<<gFF1, 256, 0, stream>>>(xn, w1b, b1, nullptr, 1.f,
                                               T_TOK, DFF, EMB, 1, 1, 0, hb, nullptr);

    // 10) out = h @ W2^T + b2 + x1     (fp32 out)
    dim3 gFF2(EMB / 128, T_TOK / 128);
    gemm_bf16_kernel<<<gFF2, 256, 0, stream>>>(hb, w2b, b2, x1, 1.f,
                                               T_TOK, EMB, DFF, 0, 0, 0, nullptr, out);
}
